// Block_32762010534337
// MI455X (gfx1250) — matrix operand order repaired and checked by reading, not compiled, not run on hardware
//
#include <hip/hip_runtime.h>

typedef __bf16 bf16;
typedef __attribute__((ext_vector_type(16))) __bf16 v16bf;
typedef __attribute__((ext_vector_type(8)))  __bf16 v8bf;
typedef __attribute__((ext_vector_type(8)))  float  v8f;

__device__ __forceinline__ bf16 f2bf(float f) {
    unsigned u = __float_as_uint(f);
    unsigned r = u + 0x7FFFu + ((u >> 16) & 1u);
    unsigned short h = (unsigned short)(r >> 16);
    return __builtin_bit_cast(bf16, h);
}

__device__ __forceinline__ v16bf combine8(v8bf lo, v8bf hi) {
    return __builtin_shufflevector(lo, hi, 0,1,2,3,4,5,6,7,8,9,10,11,12,13,14,15);
}

// Async-copy one 128x32 bf16 tile (global [row][K] slice -> LDS [row][32]).
// Per wave: exactly 2 wave-level async b128 ops (each lane moves 16B).
__device__ __forceinline__ void async_copy_tile(const bf16* gbase, unsigned ldsbase,
                                                int K, int tid) {
    #pragma unroll
    for (int c = tid; c < 512; c += 256) {
        int row = c >> 2, col = (c & 3) << 3;
        unsigned goff = (unsigned)(row * K + col) * 2u;
        unsigned loff = ldsbase + (unsigned)(row * 32 + col) * 2u;
        asm volatile("global_load_async_to_lds_b128 %0, %1, %2"
                     :: "v"(loff), "v"(goff),
                        "s"((unsigned long long)(size_t)gbase)
                     : "memory");
    }
}

// ---------------------------------------------------------------------------
// Transpose + fp32->bf16 cast:  src[K][N] (row-major) -> dst[(n+rowOff)][k]
// ---------------------------------------------------------------------------
__global__ __launch_bounds__(256)
void transpose_cast(const float* __restrict__ src, bf16* __restrict__ dst,
                    int K, int N, int rowOff) {
    __shared__ float tile[32][33];
    int k0 = blockIdx.y * 32, n0 = blockIdx.x * 32;
    int tx = threadIdx.x, ty = threadIdx.y;   // 32 x 8
    #pragma unroll
    for (int i = 0; i < 32; i += 8)
        tile[ty + i][tx] = src[(size_t)(k0 + ty + i) * N + n0 + tx];
    __syncthreads();
    #pragma unroll
    for (int i = 0; i < 32; i += 8)
        dst[(size_t)(n0 + ty + i + rowOff) * K + k0 + tx] = f2bf(tile[tx][ty + i]);
}

// ---------------------------------------------------------------------------
// Row-wise RMSNorm: x[row][C] fp32 -> out[row][C] bf16
// ---------------------------------------------------------------------------
__global__ __launch_bounds__(256)
void rmsnorm_rows(const float* __restrict__ x, bf16* __restrict__ out, int C) {
    int row = blockIdx.x;
    const float* xr = x + (size_t)row * C;
    float ssq = 0.f;
    for (int c = threadIdx.x; c < C; c += 256) { float v = xr[c]; ssq += v * v; }
    #pragma unroll
    for (int m = 16; m >= 1; m >>= 1) ssq += __shfl_xor(ssq, m, 32);
    __shared__ float red[8];
    if ((threadIdx.x & 31) == 0) red[threadIdx.x >> 5] = ssq;
    __syncthreads();
    float s = 0.f;
    #pragma unroll
    for (int i = 0; i < 8; ++i) s += red[i];
    float r = rsqrtf(s / (float)C + 1.1920929e-7f);
    bf16* orow = out + (size_t)row * C;
    for (int c = threadIdx.x; c < C; c += 256) orow[c] = f2bf(xr[c] * r);
}

// ---------------------------------------------------------------------------
// WMMA bf16 GEMM: out[M][N] = A[M][K] @ Bt[N][K]^T  (Bt stored [N][K])
// 128x128x32 tiles, 8 waves (4x2), each wave 32x64 = 2x4 wmma accums.
// Double-buffered LDS fed by global_load_async_to_lds_b128 (ASYNCcnt pipeline).
// EPI: 0 = f32 out, 1 = f32 out + residual, 2 = bf16 relu(acc)^2 out
// ---------------------------------------------------------------------------
template <int EPI>
__global__ __launch_bounds__(256)
void gemm_bf16(const bf16* __restrict__ A, const bf16* __restrict__ Bt,
               const float* __restrict__ res, float* __restrict__ outf,
               bf16* __restrict__ outb, int M, int N, int K) {
    constexpr int BM = 128, BN = 128, BK = 32;
    __shared__ __align__(64) bf16 lA[2][BM * BK];
    __shared__ __align__(64) bf16 lB[2][BN * BK];
    int tid = threadIdx.x, wid = tid >> 5, lane = tid & 31;
    int lhalf = lane >> 4, l15 = lane & 15;
    int bm = blockIdx.y * BM, bn = blockIdx.x * BN;
    int wm = (wid >> 1) * 32, wn = (wid & 1) * 64;
    v8f acc[2][4] = {};

    const bf16* Ab = A + (size_t)bm * K;
    const bf16* Bb = Bt + (size_t)bn * K;
    unsigned lAoff[2] = { (unsigned)(size_t)(void*)&lA[0][0],
                          (unsigned)(size_t)(void*)&lA[1][0] };
    unsigned lBoff[2] = { (unsigned)(size_t)(void*)&lB[0][0],
                          (unsigned)(size_t)(void*)&lB[1][0] };

    // prime the pipeline: tile 0 into buffer 0
    async_copy_tile(Ab, lAoff[0], K, tid);
    async_copy_tile(Bb, lBoff[0], K, tid);

    int nk = K / BK;
    for (int it = 0; it < nk; ++it) {
        if (it + 1 < nk) {        // stream tile it+1 while computing tile it
            int k0 = (it + 1) * BK;
            int nb = (it + 1) & 1;
            async_copy_tile(Ab + k0, lAoff[nb], K, tid);
            async_copy_tile(Bb + k0, lBoff[nb], K, tid);
            // 4 newest ops outstanding allowed; oldest 4 (tile it) must be done
            asm volatile("s_wait_asynccnt 0x4" ::: "memory");
        } else {
            asm volatile("s_wait_asynccnt 0x0" ::: "memory");
        }
        __syncthreads();

        const bf16* curA = lA[it & 1];
        const bf16* curB = lB[it & 1];
        v16bf af[2], bfr[4];
        #pragma unroll
        for (int i = 0; i < 2; ++i) {
            const bf16* p = &curA[(wm + i * 16 + l15) * BK];
            int ko = lhalf * 8;                   // CDNA5 A frag: lanes<16 K0-7,16-23
            af[i] = combine8(*(const v8bf*)&p[ko], *(const v8bf*)&p[ko + 16]);
        }
        #pragma unroll
        for (int j = 0; j < 4; ++j)               // B frag: lanes<16 K0-7,16-23 (same as A)
            bfr[j] = combine8(*(const v8bf*)&curB[(wn + j * 16 + l15) * BK + lhalf * 8],
                              *(const v8bf*)&curB[(wn + j * 16 + l15) * BK + lhalf * 8 + 16]);
        #pragma unroll
        for (int i = 0; i < 2; ++i)
            #pragma unroll
            for (int j = 0; j < 4; ++j)
                acc[i][j] = __builtin_amdgcn_wmma_f32_16x16x32_bf16(
                    false, af[i], false, bfr[j], (short)0, acc[i][j], false, false);
        __syncthreads();
    }

    int rowbase = bm + wm + lhalf * 8;            // C/D: lanes<16 M=r, >=16 M=8+r
    #pragma unroll
    for (int i = 0; i < 2; ++i)
        #pragma unroll
        for (int j = 0; j < 4; ++j) {
            int col = bn + wn + j * 16 + l15;
            #pragma unroll
            for (int r = 0; r < 8; ++r) {
                size_t idx = (size_t)(rowbase + i * 16 + r) * N + col;
                float v = acc[i][j][r];
                if (EPI == 0) outf[idx] = v;
                else if (EPI == 1) outf[idx] = v + res[idx];
                else { float t = v > 0.f ? v : 0.f; outb[idx] = f2bf(t * t); }
            }
        }
}

// ---------------------------------------------------------------------------
// Rotary + per-head RMSNorm on q/k, cast v.  qkv[row][3072] fp32 ->
// qb[row][2048], kb[row][512], vb[row][512] bf16.  One wave per head-slot.
// ---------------------------------------------------------------------------
__global__ __launch_bounds__(256)
void rope_kernel(const float* __restrict__ qkv, const float* __restrict__ cosv,
                 const float* __restrict__ sinv, bf16* __restrict__ qb,
                 bf16* __restrict__ kb, bf16* __restrict__ vb) {
    const int T = 2048;
    int row = blockIdx.x, t = row & (T - 1);
    int wid = threadIdx.x >> 5, lane = threadIdx.x & 31;
    for (int hs = wid; hs < 24; hs += 8) {
        const float* src; bf16* dst; bool dorope;
        if (hs < 16)      { src = qkv + (size_t)row*3072 + hs*128;            dst = qb + (size_t)row*2048 + hs*128;      dorope = true; }
        else if (hs < 20) { int h = hs-16; src = qkv + (size_t)row*3072 + 2048 + h*128; dst = kb + (size_t)row*512 + h*128; dorope = true; }
        else              { int h = hs-20; src = qkv + (size_t)row*3072 + 2560 + h*128; dst = vb + (size_t)row*512 + h*128; dorope = false; }
        if (dorope) {
            float r1[2], r2[2], ssq = 0.f;
            #pragma unroll
            for (int i = 0; i < 2; ++i) {
                int d = lane + i * 32;
                float c = cosv[t * 64 + d], s = sinv[t * 64 + d];
                float x1 = src[d], x2 = src[d + 64];
                r1[i] = x1 * c + x2 * s;
                r2[i] = -x1 * s + x2 * c;
                ssq += r1[i] * r1[i] + r2[i] * r2[i];
            }
            #pragma unroll
            for (int m = 1; m < 32; m <<= 1) ssq += __shfl_xor(ssq, m, 32);
            float rms = rsqrtf(ssq * (1.f / 128.f) + 1.1920929e-7f);
            #pragma unroll
            for (int i = 0; i < 2; ++i) {
                int d = lane + i * 32;
                dst[d]      = f2bf(r1[i] * rms);
                dst[d + 64] = f2bf(r2[i] * rms);
            }
        } else {
            for (int d = lane; d < 128; d += 32) dst[d] = f2bf(src[d]);
        }
    }
}

// ---------------------------------------------------------------------------
// Flash attention (causal, GQA 4:1) with WMMA. Block = (b, h, 128 queries),
// 8 waves x 16-query tiles; 32-key tiles staged in LDS.
// Q:[B,T,16,128] K,V:[B,T,4,128] bf16 -> Y:[B,T,2048] bf16
// ---------------------------------------------------------------------------
__global__ __launch_bounds__(256)
void flash_attn(const bf16* __restrict__ Q, const bf16* __restrict__ Kg,
                const bf16* __restrict__ V, bf16* __restrict__ Y) {
    const int T = 2048, NH = 16, NKV = 4, HD = 128;
    int bq = blockIdx.x, h = blockIdx.y, b = blockIdx.z;
    int hk = h >> 2;
    int tid = threadIdx.x, wid = tid >> 5, lane = tid & 31;
    int lhalf = lane >> 4, l15 = lane & 15;
    int q0 = bq * 128 + wid * 16;
    __shared__ __align__(64) bf16 lK[32 * 128];   // [key][ch]
    __shared__ __align__(64) bf16 lVT[128 * 32];  // [ch][key]
    __shared__ __align__(64) bf16 lP[8][16 * 32]; // per-wave P repack

    v16bf qf[4];
    {
        const bf16* qrow = Q + (((size_t)b * T + q0 + l15) * NH + h) * HD;
        int ko = lhalf * 8;
        #pragma unroll
        for (int c = 0; c < 4; ++c)
            qf[c] = combine8(*(const v8bf*)&qrow[c * 32 + ko],
                             *(const v8bf*)&qrow[c * 32 + ko + 16]);
    }
    v8f o[8] = {};
    float m_[8], l_[8];
    #pragma unroll
    for (int r = 0; r < 8; ++r) { m_[r] = -1e30f; l_[r] = 0.f; }
    const float scale = 0.08838834764831845f;  // 1/sqrt(128)
    int nkt = bq * 4 + 4;

    for (int kt = 0; kt < nkt; ++kt) {
        int kbase = kt * 32;
        #pragma unroll
        for (int c = tid; c < 512; c += 256) {    // K tile: 32x128
            int key = c >> 4, col = (c & 15) << 3;
            *(v8bf*)&lK[key * 128 + col] =
                *(const v8bf*)&Kg[(((size_t)b * T + kbase + key) * NKV + hk) * HD + col];
        }
        #pragma unroll
        for (int c = tid; c < 512; c += 256) {    // V tile transposed
            int key = c >> 4, col = (c & 15) << 3;
            v8bf vv = *(const v8bf*)&V[(((size_t)b * T + kbase + key) * NKV + hk) * HD + col];
            #pragma unroll
            for (int e = 0; e < 8; ++e) lVT[(col + e) * 32 + key] = vv[e];
        }
        __syncthreads();
        if (kbase <= q0 + 15) {                   // wave-uniform
            v8f s[2] = {};
            #pragma unroll
            for (int st = 0; st < 2; ++st)
                #pragma unroll
                for (int c = 0; c < 4; ++c) {
                    v16bf kf = combine8(*(const v8bf*)&lK[(st * 16 + l15) * 128 + c * 32 + lhalf * 8],
                                        *(const v8bf*)&lK[(st * 16 + l15) * 128 + c * 32 + lhalf * 8 + 16]);
                    s[st] = __builtin_amdgcn_wmma_f32_16x16x32_bf16(
                        false, qf[c], false, kf, (short)0, s[st], false, false);
                }
            int qlo = q0 + lhalf * 8;
            float rmax[8], rs[8];
            #pragma unroll
            for (int r = 0; r < 8; ++r) {
                int qi = qlo + r;
                float best = -1e30f;
                #pragma unroll
                for (int st = 0; st < 2; ++st) {
                    int ki = kbase + st * 16 + l15;
                    float v = s[st][r] * scale;
                    if (ki > qi) v = -1e30f;
                    s[st][r] = v;
                    best = fmaxf(best, v);
                }
                rmax[r] = best;
            }
            #pragma unroll
            for (int r = 0; r < 8; ++r) {         // row-max over the 16-lane group
                float v = rmax[r];
                #pragma unroll
                for (int msk = 1; msk < 16; msk <<= 1) v = fmaxf(v, __shfl_xor(v, msk, 16));
                rmax[r] = v;
            }
            #pragma unroll
            for (int r = 0; r < 8; ++r) {
                float mn = fmaxf(m_[r], rmax[r]);
                float f = __expf(m_[r] - mn);
                float sum = 0.f;
                #pragma unroll
                for (int st = 0; st < 2; ++st) {
                    float p = __expf(s[st][r] - mn);
                    s[st][r] = p; sum += p;
                }
                rs[r] = sum; m_[r] = mn; l_[r] *= f;
                #pragma unroll
                for (int n = 0; n < 8; ++n) o[n][r] *= f;
            }
            #pragma unroll
            for (int r = 0; r < 8; ++r) {         // row-sum
                float v = rs[r];
                #pragma unroll
                for (int msk = 1; msk < 16; msk <<= 1) v += __shfl_xor(v, msk, 16);
                l_[r] += v;
            }
            bf16* pw = lP[wid];                   // repack C-layout -> A-layout
            #pragma unroll
            for (int st = 0; st < 2; ++st)
                #pragma unroll
                for (int r = 0; r < 8; ++r)
                    pw[(r + lhalf * 8) * 32 + st * 16 + l15] = f2bf(s[st][r]);
            asm volatile("s_wait_dscnt 0" ::: "memory");
            v16bf pa;
            {
                const bf16* pr = &pw[l15 * 32];
                int ko = lhalf * 8;
                pa = combine8(*(const v8bf*)&pr[ko], *(const v8bf*)&pr[ko + 16]);
            }
            #pragma unroll
            for (int n = 0; n < 8; ++n) {
                v16bf vf = combine8(*(const v8bf*)&lVT[(n * 16 + l15) * 32 + lhalf * 8],
                                    *(const v8bf*)&lVT[(n * 16 + l15) * 32 + lhalf * 8 + 16]);
                o[n] = __builtin_amdgcn_wmma_f32_16x16x32_bf16(
                    false, pa, false, vf, (short)0, o[n], false, false);
            }
        }
        __syncthreads();
    }
    bf16* yb = Y + (size_t)b * T * 2048;
    int qlo = q0 + lhalf * 8;
    #pragma unroll
    for (int r = 0; r < 8; ++r) {
        float inv = 1.f / l_[r];
        #pragma unroll
        for (int n = 0; n < 8; ++n)
            yb[(size_t)(qlo + r) * 2048 + h * 128 + n * 16 + l15] = f2bf(o[n][r] * inv);
    }
}

// ---------------------------------------------------------------------------
extern "C" void kernel_launch(void* const* d_in, const int* in_sizes, int n_in,
                              void* d_out, int out_size, void* d_ws, size_t ws_size,
                              hipStream_t stream) {
    (void)in_sizes; (void)n_in; (void)out_size; (void)ws_size;
    const int T = 2048, C = 2048, DFF = 8192;
    const int M = 2 * T;  // 4096 rows

    const float* x     = (const float*)d_in[0];
    const float* cosv  = (const float*)d_in[1];
    const float* sinv  = (const float*)d_in[2];
    const float* wq    = (const float*)d_in[3];
    const float* wk    = (const float*)d_in[4];
    const float* wv    = (const float*)d_in[5];
    const float* wo    = (const float*)d_in[6];
    const float* wfc   = (const float*)d_in[7];
    const float* wproj = (const float*)d_in[8];
    float* out = (float*)d_out;

    char* ws = (char*)d_ws; size_t off = 0;
    auto alloc = [&](size_t bytes) { void* p = ws + off; off += (bytes + 255) & ~(size_t)255; return p; };
    bf16*  wqkvT  = (bf16*)alloc((size_t)3072 * 2048 * 2);
    bf16*  woT    = (bf16*)alloc((size_t)2048 * 2048 * 2);
    bf16*  wfcT   = (bf16*)alloc((size_t)8192 * 2048 * 2);
    bf16*  wprojT = (bf16*)alloc((size_t)2048 * 8192 * 2);
    bf16*  hbf    = (bf16*)alloc((size_t)M * 2048 * 2);        // reused: h, y, h2
    void*  big    = alloc((size_t)M * 8192 * 2);               // qkv f32 then ffn bf16
    float* qkv    = (float*)big;
    bf16*  ffnb   = (bf16*)big;
    bf16*  qb     = (bf16*)alloc((size_t)M * 2048 * 2);
    bf16*  kb     = (bf16*)alloc((size_t)M * 512 * 2);
    bf16*  vb     = (bf16*)alloc((size_t)M * 512 * 2);
    float* x1     = (float*)alloc((size_t)M * 2048 * 4);

    dim3 tb(32, 8);
    // Weight transposes into bf16 [N][K]
    transpose_cast<<<dim3(2048/32, 2048/32), tb, 0, stream>>>(wq,    wqkvT,  2048, 2048, 0);
    transpose_cast<<<dim3( 512/32, 2048/32), tb, 0, stream>>>(wk,    wqkvT,  2048,  512, 2048);
    transpose_cast<<<dim3( 512/32, 2048/32), tb, 0, stream>>>(wv,    wqkvT,  2048,  512, 2560);
    transpose_cast<<<dim3(2048/32, 2048/32), tb, 0, stream>>>(wo,    woT,    2048, 2048, 0);
    transpose_cast<<<dim3(8192/32, 2048/32), tb, 0, stream>>>(wfc,   wfcT,   2048, 8192, 0);
    transpose_cast<<<dim3(2048/32, 8192/32), tb, 0, stream>>>(wproj, wprojT, 8192, 2048, 0);

    // h = rmsnorm(x)
    rmsnorm_rows<<<M, 256, 0, stream>>>(x, hbf, C);
    // qkv = h @ [wq|wk|wv]
    gemm_bf16<0><<<dim3(3072/128, M/128), 256, 0, stream>>>(hbf, wqkvT, nullptr, qkv, nullptr, M, 3072, C);
    // rotary + head-rmsnorm (q,k), cast (v)
    rope_kernel<<<M, 256, 0, stream>>>(qkv, cosv, sinv, qb, kb, vb);
    // flash attention -> y (reuse hbf)
    flash_attn<<<dim3(T/128, 16, 2), 256, 0, stream>>>(qb, kb, vb, hbf);
    // x1 = x + y @ wo
    gemm_bf16<1><<<dim3(2048/128, M/128), 256, 0, stream>>>(hbf, woT, x, x1, nullptr, M, 2048, C);
    // h2 = rmsnorm(x1) (reuse hbf)
    rmsnorm_rows<<<M, 256, 0, stream>>>(x1, hbf, C);
    // ffn = relu(h2 @ w_fc)^2 (bf16)
    gemm_bf16<2><<<dim3(8192/128, M/128), 256, 0, stream>>>(hbf, wfcT, nullptr, nullptr, ffnb, M, DFF, C);
    // out = x1 + ffn @ w_proj
    gemm_bf16<1><<<dim3(2048/128, M/128), 256, 0, stream>>>(ffnb, wprojT, x1, out, nullptr, M, 2048, DFF);
}
